// ROIAttentionPool_29454885716446
// MI455X (gfx1250) — matrix operand order repaired and checked by reading, not compiled, not run on hardware
//
#include <hip/hip_runtime.h>
#include <hip/hip_fp16.h>

typedef __attribute__((ext_vector_type(16))) _Float16 v16h;
typedef __attribute__((ext_vector_type(8)))  float    v8f;
typedef unsigned int v4u __attribute__((ext_vector_type(4)));
typedef int          v4i __attribute__((ext_vector_type(4)));
typedef int          v8i __attribute__((ext_vector_type(8)));

#define BATCH   16
#define NPTS    16384
#define DIM     64
#define FEAT    768
#define NQ      4
#define RADIUS  0.15f
#define NEGINF  (-1e9f)
#define NBLK    8            // streaming blocks per batch
#define CHUNK   128          // points per LDS chunk
#define NCHUNK  (NPTS / (NBLK * CHUNK))   // 16 chunks per block
#define CHUNKB  (CHUNK * DIM * 4)         // bytes per staged chunk (32 KB)

#if defined(__has_builtin)
#  if __has_builtin(__builtin_amdgcn_tensor_load_to_lds)
#    define HAVE_TDM 1
#  endif
#endif
#ifndef HAVE_TDM
#  define HAVE_TDM 0
#endif

#if HAVE_TDM
// ---------------------------------------------------------------------------
// Issue one TDM copy: 32 KB contiguous (128 pts x 64 dims fp32) global -> LDS.
// D# group0: count=1 | lds_addr | global_addr[56:0] | type=2
// D# group1: data_size=4B, tensor_dim0=tile_dim0=8192, dim1=1, stride0=8192
// ---------------------------------------------------------------------------
__device__ __forceinline__ void tdm_load_chunk(const float* gsrc, unsigned lds_off) {
    const unsigned long long ga = reinterpret_cast<unsigned long long>(gsrc);
    v4u g0 = { 1u,                                    // count = 1 valid descriptor
               lds_off,                               // LDS byte address
               (unsigned)(ga & 0xffffffffu),          // global_addr[31:0]
               (unsigned)((ga >> 32) & 0x01ffffffu) | 0x80000000u }; // ga[56:32] | type=2
    v8i g1 = { (int)0x00020000,   // data_size = 2 (4 bytes), no flags
               (int)0x20000000,   // tensor_dim0[15:0]=8192 in bits[31:16]
               (int)0x00010000,   // tensor_dim0[31:16]=0, tensor_dim1[15:0]=1
               (int)0x20000000,   // tensor_dim1[31:16]=0, tile_dim0=8192
               (int)0x00000001,   // tile_dim1=1, tile_dim2=0
               (int)0x00002000,   // tensor_dim0_stride = 8192
               0, 0 };            // stride0 hi / tensor_dim1_stride = 0
    v4i gz = { 0, 0, 0, 0 };
#if __clang_major__ >= 23
    v8i gz8 = { 0, 0, 0, 0, 0, 0, 0, 0 };
    __builtin_amdgcn_tensor_load_to_lds(g0, g1, gz, gz, gz8, 0);
#else
    __builtin_amdgcn_tensor_load_to_lds(g0, g1, gz, gz, 0);
#endif
}
#endif

// ---------------------------------------------------------------------------
// K0: init per-batch min-multiplier
// ---------------------------------------------------------------------------
__global__ void roi_init_kernel(int* __restrict__ minm) {
    if (threadIdx.x < BATCH) minm[threadIdx.x] = 0x3fffffff;
}

// ---------------------------------------------------------------------------
// K1: Qp[q][d] = sum_f qt[q][f] * W[d][f];  qb[q] = qt[q] . b_proj
// ---------------------------------------------------------------------------
__global__ void __launch_bounds__(256)
roi_qp_kernel(const float* __restrict__ qt, const float* __restrict__ W,
              const float* __restrict__ bias,
              float* __restrict__ Qp, float* __restrict__ qb) {
    const int tid = threadIdx.x;
    const int q = tid >> 6, d = tid & 63;
    float s = 0.f;
    for (int f = 0; f < FEAT; ++f)
        s = fmaf(qt[q * FEAT + f], W[d * FEAT + f], s);
    Qp[q * DIM + d] = s;
    if (tid < NQ) {
        float t = 0.f;
        for (int f = 0; f < FEAT; ++f) t = fmaf(qt[tid * FEAT + f], bias[f], t);
        qb[tid] = t;
    }
}

// ---------------------------------------------------------------------------
// K2: per-batch smallest radius multiplier (wave-reduced atomicMin)
// ---------------------------------------------------------------------------
__global__ void __launch_bounds__(256)
roi_radius_kernel(const float* __restrict__ xyz, const float* __restrict__ poi,
                  int* __restrict__ minm) {
    const int i = blockIdx.x * blockDim.x + threadIdx.x;     // 0..B*N-1
    const int b = i / NPTS;
    const float dx = xyz[(size_t)i * 3 + 0] - poi[b * 3 + 0];
    const float dy = xyz[(size_t)i * 3 + 1] - poi[b * 3 + 1];
    const float dz = xyz[(size_t)i * 3 + 2] - poi[b * 3 + 2];
    const float dst = sqrtf(dx * dx + dy * dy + dz * dz);
    int m = (int)floorf(dst / RADIUS) + 1;                   // smallest r with dist < 0.15*r
#pragma unroll
    for (int off = 16; off >= 1; off >>= 1)
        m = min(m, __shfl_xor(m, off, 32));
    if ((threadIdx.x & 31) == 0) atomicMin(&minm[b], m);
}

// ---------------------------------------------------------------------------
// K3: streaming pass — TDM double-buffered staging, WMMA scores,
//     online-softmax pooling in 64-dim space.
// grid: (NBLK, BATCH), block: 256 threads (8 waves)
// LDS: 2 x 32 KB fp32 tile (double buffer) + 2 KB scores  (> 64 KB: WGP LDS)
// ---------------------------------------------------------------------------
__global__ void __launch_bounds__(256)
roi_stream_kernel(const float* __restrict__ xyz, const float* __restrict__ feat,
                  const float* __restrict__ poi, const float* __restrict__ Qp,
                  const float* __restrict__ qb, const int* __restrict__ minm,
                  float* __restrict__ part) {
    __shared__ float fh32[2 * CHUNK * DIM];   // 64 KB double-buffered feature tile
    __shared__ float sc[CHUNK * NQ];          // scores -> weights (also end reduce)
    __shared__ float Mq[NQ];                  // running max per query
    __shared__ float scq[NQ];                 // per-chunk rescale factor

    const int tid  = threadIdx.x;
    const int lane = tid & 31;
    const int wv   = tid >> 5;
    const int b    = blockIdx.y;
    const int blk  = blockIdx.x;
    const int pbase = blk * (NCHUNK * CHUNK); // this block's point range base

    const int   mm  = minm[b];
    const float er  = RADIUS * ((mm <= 9) ? (float)mm : 1.0f);
    const float er2 = er * er;
    const float px = poi[b * 3 + 0], py = poi[b * 3 + 1], pz = poi[b * 3 + 2];
    const float qb0 = qb[0], qb1 = qb[1], qb2 = qb[2], qb3 = qb[3];

    // A-matrix: Qp padded 4->16 rows, f16, two K-chunks (K=0..31, 32..63)
    const int am = lane & 15;
    const int ah = lane >> 4;
    v16h a0, a1;
#pragma unroll
    for (int e = 0; e < 16; ++e) {
        const int v  = e >> 1;
        const int kk = 2 * v + (e & 1) + 16 * ah;
        a0[e] = (_Float16)((am < NQ) ? Qp[am * DIM + kk]      : 0.0f);
        a1[e] = (_Float16)((am < NQ) ? Qp[am * DIM + 32 + kk] : 0.0f);
    }

    const int qi = tid >> 6;     // owned query
    const int di = tid & 63;     // owned dim
    float aAcc = 0.0f;           // numerator accumulator A[q][d]
    float dAcc = 0.0f;           // denominator partial

    if (tid < NQ) Mq[tid] = -1e30f;

#if HAVE_TDM
    const unsigned fhBase = (unsigned)reinterpret_cast<unsigned long long>(&fh32[0]);
    if (wv == 0)   // prologue: fetch chunk 0 into buffer 0 (async, TENSORcnt)
        tdm_load_chunk(feat + (size_t)(b * NPTS + pbase) * DIM, fhBase);
#endif
    __syncthreads();

    for (int ck = 0; ck < NCHUNK; ++ck) {
        const int base = pbase + ck * CHUNK;
        float* fb = fh32 + (ck & 1) * (CHUNK * DIM);

#if HAVE_TDM
        // wave 0: launch next chunk into the other buffer, then wait so that
        // the *current* buffer (older TDM op, in-order) is complete.
        if (wv == 0) {
            if (ck + 1 < NCHUNK) {
                tdm_load_chunk(feat + (size_t)(b * NPTS + base + CHUNK) * DIM,
                               fhBase + (unsigned)(((ck + 1) & 1) * CHUNKB));
                __builtin_amdgcn_s_wait_tensorcnt(1);
            } else {
                __builtin_amdgcn_s_wait_tensorcnt(0);
            }
        }
        __syncthreads();   // publish TDM-written LDS to all waves
#else
        // fallback: manual coalesced staging + next-chunk prefetch
        {
            const float4* src =
                reinterpret_cast<const float4*>(feat + (size_t)(b * NPTS + base) * DIM);
            float4* dst = reinterpret_cast<float4*>(fb);
#pragma unroll
            for (int j = 0; j < 8; ++j) dst[tid + j * 256] = src[tid + j * 256];
            if (ck + 1 < NCHUNK) {
                const char* pf =
                    (const char*)(feat + (size_t)(b * NPTS + base + CHUNK) * DIM) + tid * 128;
                __builtin_prefetch(pf, 0, 1);
            }
        }
        __syncthreads();
#endif

        // ---- scores: each wave does one 16-point tile, 2 x v_wmma_f32_16x16x32_f16
        {
            const int nt = wv * 16;
            const int bn = nt + (lane & 15);
            v16h b0, b1;
#pragma unroll
            for (int e = 0; e < 16; ++e) {
                const int v  = e >> 1;
                const int kb = 16 * ah + 2 * v + (e & 1);
                b0[e] = (_Float16)fb[bn * DIM + kb];
                b1[e] = (_Float16)fb[bn * DIM + 32 + kb];
            }
            v8f c = {};
            c = __builtin_amdgcn_wmma_f32_16x16x32_f16(false, a0, false, b0,
                                                       (short)0, c, false, false);
            c = __builtin_amdgcn_wmma_f32_16x16x32_f16(false, a1, false, b1,
                                                       (short)0, c, false, false);
            if (lane < 16) {                 // rows 0..3 of the tile are live queries
                const int n  = nt + lane;
                const int gn = b * NPTS + base + n;
                const float dx = xyz[gn * 3 + 0] - px;
                const float dy = xyz[gn * 3 + 1] - py;
                const float dz = xyz[gn * 3 + 2] - pz;
                const bool in = (dx * dx + dy * dy + dz * dz) < er2;
                sc[n * NQ + 0] = in ? (c[0] + qb0) : NEGINF;
                sc[n * NQ + 1] = in ? (c[1] + qb1) : NEGINF;
                sc[n * NQ + 2] = in ? (c[2] + qb2) : NEGINF;
                sc[n * NQ + 3] = in ? (c[3] + qb3) : NEGINF;
            }
        }
        __syncthreads();

        // ---- per-chunk max + running-max merge (waves 0..3, one per query)
        if (tid < 128) {
            const int q = wv;
            float m = sc[lane * NQ + q];
            m = fmaxf(m, sc[(lane + 32) * NQ + q]);
            m = fmaxf(m, sc[(lane + 64) * NQ + q]);
            m = fmaxf(m, sc[(lane + 96) * NQ + q]);
#pragma unroll
            for (int off = 16; off >= 1; off >>= 1)
                m = fmaxf(m, __shfl_xor(m, off, 32));
            if (lane == 0) {
                const float mo = Mq[q];
                const float mn = fmaxf(mo, m);
                scq[q] = __expf(mo - mn);
                Mq[q]  = mn;
            }
        }
        __syncthreads();

        // ---- weights = exp(s - M) ----
        if (tid < CHUNK) {
#pragma unroll
            for (int q = 0; q < NQ; ++q)
                sc[tid * NQ + q] = __expf(sc[tid * NQ + q] - Mq[q]);
        }
        __syncthreads();

        // ---- rescale + accumulate numerator/denominator (fp32 LDS reads) ----
        {
            const float s = scq[qi];
            aAcc *= s; dAcc *= s;
            dAcc += sc[di * NQ + qi] + sc[(di + 64) * NQ + qi];
#pragma unroll 8
            for (int n = 0; n < CHUNK; ++n)
                aAcc = fmaf(sc[n * NQ + qi], fb[n * DIM + di], aAcc);
        }
        __syncthreads();
    }

    // ---- write block partials: [M, D, A[64]] per (b, blk, q) ----
    sc[tid] = dAcc;              // reuse score buffer as reduction scratch
    __syncthreads();
    float* pb = part + (((size_t)(b * NBLK + blk) * NQ + qi) * 66);
    pb[2 + di] = aAcc;
    if (tid < NQ) {
        float Dq = 0.0f;
        for (int j = 0; j < 64; ++j) Dq += sc[tid * 64 + j];
        float* p0 = part + (((size_t)(b * NBLK + blk) * NQ + tid) * 66);
        p0[0] = Mq[tid];
        p0[1] = Dq;
    }
}

// ---------------------------------------------------------------------------
// K4: log-sum-exp merge of block partials -> pooled features (B, Q, 64)
// ---------------------------------------------------------------------------
__global__ void __launch_bounds__(256)
roi_merge_kernel(const float* __restrict__ part, const int* __restrict__ minm,
                 float* __restrict__ pfeat) {
    const int b = blockIdx.x;
    const int tid = threadIdx.x;
    const int q = tid >> 6, d = tid & 63;
    float Mg = -1e30f;
    for (int k = 0; k < NBLK; ++k)
        Mg = fmaxf(Mg, part[(((size_t)(b * NBLK + k) * NQ + q) * 66)]);
    float A = 0.f, D = 0.f;
    for (int k = 0; k < NBLK; ++k) {
        const float* p = part + (((size_t)(b * NBLK + k) * NQ + q) * 66);
        const float e = __expf(p[0] - Mg);
        D += p[1] * e;
        A += p[2 + d] * e;
    }
    const bool hasAny = (minm[b] <= 9);
    pfeat[(size_t)(b * NQ + q) * DIM + d] = (hasAny && D > 0.f) ? (A / D) : 0.0f;
}

// ---------------------------------------------------------------------------
// K5: pooled = pfeat(64x64) @ W(64x768) + bias, via WMMA; has_any zeroing fused
// grid: (6, 4), block 256 (8 waves, one 16x16 output tile each)
// ---------------------------------------------------------------------------
__global__ void __launch_bounds__(256)
roi_proj_wmma_kernel(const float* __restrict__ pfeat, const float* __restrict__ W,
                     const float* __restrict__ bias, const int* __restrict__ minm,
                     float* __restrict__ pool) {
    const int tid  = threadIdx.x;
    const int lane = tid & 31;
    const int wv   = tid >> 5;
    const int nt   = blockIdx.x * 8 + wv;   // 0..47
    const int mt   = blockIdx.y;            // 0..3
    const int am   = lane & 15;
    const int ah   = lane >> 4;
    const int col  = nt * 16 + am;

    v16h a0, a1, b0, b1;
#pragma unroll
    for (int e = 0; e < 16; ++e) {
        const int v  = e >> 1;
        const int ka = 2 * v + (e & 1) + 16 * ah;
        const int row = mt * 16 + am;
        a0[e] = (_Float16)pfeat[row * DIM + ka];
        a1[e] = (_Float16)pfeat[row * DIM + 32 + ka];
        const int kb = 16 * ah + 2 * v + (e & 1);
        b0[e] = (_Float16)W[kb * FEAT + col];
        b1[e] = (_Float16)W[(32 + kb) * FEAT + col];
    }
    v8f c = {};
    c = __builtin_amdgcn_wmma_f32_16x16x32_f16(false, a0, false, b0, (short)0, c, false, false);
    c = __builtin_amdgcn_wmma_f32_16x16x32_f16(false, a1, false, b1, (short)0, c, false, false);
#pragma unroll
    for (int r = 0; r < 8; ++r) {
        const int row = mt * 16 + r + 8 * ah;     // row = b*NQ + q
        float vv = c[r] + bias[col];
        if (minm[row >> 2] > 9) vv = 0.0f;        // !has_any -> pooled = 0
        pool[(size_t)row * FEAT + col] = vv;
    }
}

// ---------------------------------------------------------------------------
// K6: LayerNorm over FEAT per (b, q) row -> d_out
// ---------------------------------------------------------------------------
__global__ void __launch_bounds__(256)
roi_ln_kernel(const float* __restrict__ pool, const float* __restrict__ gamma,
              const float* __restrict__ beta, float* __restrict__ out) {
    __shared__ float s1[256], s2[256];
    const int row = blockIdx.x, tid = threadIdx.x;
    float a = 0.f, b2 = 0.f;
#pragma unroll
    for (int j = 0; j < 3; ++j) {
        const float x = pool[(size_t)row * FEAT + tid + j * 256];
        a += x; b2 += x * x;
    }
    s1[tid] = a; s2[tid] = b2;
    __syncthreads();
    for (int off = 128; off >= 1; off >>= 1) {
        if (tid < off) { s1[tid] += s1[tid + off]; s2[tid] += s2[tid + off]; }
        __syncthreads();
    }
    const float mu  = s1[0] / (float)FEAT;
    const float var = s2[0] / (float)FEAT - mu * mu;
    const float rs  = rsqrtf(fmaxf(var, 0.f) + 1e-5f);
#pragma unroll
    for (int j = 0; j < 3; ++j) {
        const int f = tid + j * 256;
        const float x = pool[(size_t)row * FEAT + f];
        out[(size_t)row * FEAT + f] = (x - mu) * rs * gamma[f] + beta[f];
    }
}

// ---------------------------------------------------------------------------
extern "C" void kernel_launch(void* const* d_in, const int* in_sizes, int n_in,
                              void* d_out, int out_size, void* d_ws, size_t ws_size,
                              hipStream_t stream) {
    (void)in_sizes; (void)n_in; (void)out_size; (void)ws_size;
    const float* xyz   = (const float*)d_in[0];
    const float* feat  = (const float*)d_in[1];
    const float* poi   = (const float*)d_in[2];
    const float* qt    = (const float*)d_in[3];
    const float* W     = (const float*)d_in[4];
    const float* bias  = (const float*)d_in[5];
    const float* gamma = (const float*)d_in[6];
    const float* beta  = (const float*)d_in[7];
    // d_in[8] = ds (==1, identity downsample) — unused

    float* wsf   = (float*)d_ws;
    float* Qp    = wsf;                     // 256 f
    float* qb    = wsf + 256;               // 4 f
    int*   minm  = (int*)(wsf + 272);       // 16 i32
    float* part  = wsf + 512;               // 16*8*4*66 = 33792 f
    float* pfeat = wsf + 34304;             // 16*4*64  = 4096 f
    float* pool  = wsf + 38400;             // 64*768   = 49152 f
    float* out   = (float*)d_out;

    roi_init_kernel<<<1, 32, 0, stream>>>(minm);
    roi_qp_kernel<<<1, 256, 0, stream>>>(qt, W, bias, Qp, qb);
    roi_radius_kernel<<<(BATCH * NPTS) / 256, 256, 0, stream>>>(xyz, poi, minm);
    roi_stream_kernel<<<dim3(NBLK, BATCH), 256, 0, stream>>>(xyz, feat, poi, Qp, qb,
                                                             minm, part);
    roi_merge_kernel<<<BATCH, 256, 0, stream>>>(part, minm, pfeat);
    roi_proj_wmma_kernel<<<dim3(6, 4), 256, 0, stream>>>(pfeat, W, bias, minm, pool);
    roi_ln_kernel<<<BATCH * NQ, 256, 0, stream>>>(pool, gamma, beta, out);
}
